// LorentzGNN_73710228733975
// MI455X (gfx1250) — matrix operand order repaired and checked by reading, not compiled, not run on hardware
//
#include <hip/hip_runtime.h>
#include <hip/hip_bf16.h>

// ---------------------------------------------------------------------------
// LorentzGNN for MI455X (gfx1250): all GEMMs via v_wmma_f32_16x16x32_bf16,
// edge softmax via ordered-uint atomicMax (deterministic) + f32 atomics.
// Whole working set (~160 MB) is L2-resident on MI455X (192 MB L2).
// ---------------------------------------------------------------------------
#define NNODE 16384
#define NEDGE 131072
#define NETOT (NNODE + NEDGE)
#define HID   512
#define BATCH 128

typedef __attribute__((ext_vector_type(16))) __bf16 v16bf;
typedef __attribute__((ext_vector_type(8)))  __bf16 v8bf;
typedef __attribute__((ext_vector_type(8)))  float  v8f;

union ABfrag { v16bf v; v8bf h[2]; };

__device__ __forceinline__ float geluf(float x) {
    return 0.5f * x * (1.0f + erff(x * 0.70710678118654752440f));
}
__device__ __forceinline__ unsigned orderKey(float f) {
    unsigned b = __float_as_uint(f);
    return (b & 0x80000000u) ? ~b : (b | 0x80000000u);
}
__device__ __forceinline__ float keyToFloat(unsigned k) {
    unsigned b = (k & 0x80000000u) ? (k & 0x7FFFFFFFu) : ~k;
    return __uint_as_float(b);
}

// C[M x ldc] = A[M x K](bf16) * BT[Npad x K](bf16)^T + bias[0..nbias)
// K%32==0, M%128==0, Npad = grid.x*128, ldc = Npad. Wave tile 32x64.
__global__ __launch_bounds__(256) void gemm_bf16_kernel(
    const __bf16* __restrict__ A, const __bf16* __restrict__ BT,
    const float* __restrict__ bias, int nbias,
    float* __restrict__ C, int K, int lda, int ldb, int ldc) {

    const int lane = threadIdx.x & 31;
    const int wid  = threadIdx.x >> 5;
    const int m0   = blockIdx.y * 128 + (wid & 3) * 32;
    const int n0   = blockIdx.x * 128 + (wid >> 2) * 64;
    const int sub  = lane & 15;
    const int half = lane >> 4;

    v8f acc[2][4];
#pragma unroll
    for (int i = 0; i < 2; ++i)
#pragma unroll
        for (int j = 0; j < 4; ++j)
#pragma unroll
            for (int r = 0; r < 8; ++r) acc[i][j][r] = 0.0f;

    for (int k0 = 0; k0 < K; k0 += 32) {
        ABfrag a[2];
#pragma unroll
        for (int i = 0; i < 2; ++i) {
            const __bf16* p = A + (size_t)(m0 + i * 16 + sub) * lda + k0 + half * 8;
            a[i].h[0] = *(const v8bf*)(p);
            a[i].h[1] = *(const v8bf*)(p + 16);
        }
        ABfrag b[4];
#pragma unroll
        for (int j = 0; j < 4; ++j) {
            const __bf16* p = BT + (size_t)(n0 + j * 16 + sub) * ldb + k0 + half * 8;
            b[j].h[0] = *(const v8bf*)(p);
            b[j].h[1] = *(const v8bf*)(p + 16);
        }
#pragma unroll
        for (int i = 0; i < 2; ++i)
#pragma unroll
            for (int j = 0; j < 4; ++j)
                acc[i][j] = __builtin_amdgcn_wmma_f32_16x16x32_bf16(
                    false, a[i].v, false, b[j].v, (short)0, acc[i][j], false, false);
    }

#pragma unroll
    for (int i = 0; i < 2; ++i) {
#pragma unroll
        for (int j = 0; j < 4; ++j) {
            const int col = n0 + j * 16 + sub;
            const float bv = (col < nbias) ? bias[col] : 0.0f;
#pragma unroll
            for (int r = 0; r < 8; ++r) {
                const int row = m0 + i * 16 + r + half * 8;
                C[(size_t)row * ldc + col] = acc[i][j][r] + bv;
            }
        }
    }
}

__global__ void cast_x_kernel(const float* __restrict__ x, __bf16* __restrict__ hbf) {
    int idx = blockIdx.x * blockDim.x + threadIdx.x;
    if (idx >= NNODE * HID) return;
    int n = idx >> 9, k = idx & 511;
    hbf[idx] = (__bf16)x[(size_t)n * 513 + 1 + k];
}

__global__ void gelu_cast_kernel(const float* __restrict__ h, __bf16* __restrict__ hbf) {
    int idx = blockIdx.x * blockDim.x + threadIdx.x;
    if (idx >= NNODE * HID) return;
    hbf[idx] = (__bf16)geluf(h[idx]);
}

// W (K x N, f32) -> WT (Np x Kp, bf16, zero padded)
__global__ void transpose_cast_kernel(const float* __restrict__ W, int K, int N,
                                      __bf16* __restrict__ WT, int Kp, int Np) {
    int idx = blockIdx.x * blockDim.x + threadIdx.x;
    if (idx >= Np * Kp) return;
    int np = idx / Kp, kp = idx - np * Kp;
    float v = (np < N && kp < K) ? W[(size_t)kp * N + np] : 0.0f;
    WT[idx] = (__bf16)v;
}

__global__ void bias_init_kernel(const float* __restrict__ bias, float* __restrict__ out) {
    int idx = blockIdx.x * blockDim.x + threadIdx.x;
    if (idx >= NNODE * HID) return;
    out[idx] = bias[idx & 511];
}

__global__ void edge_logits_kernel(const int* __restrict__ ei,
                                   const float* __restrict__ xl,
                                   const float* __restrict__ xr,
                                   const float* __restrict__ att,
                                   float* __restrict__ elog,
                                   unsigned* __restrict__ emaxU) {
    int e = blockIdx.x;
    int lane = threadIdx.x & 31;
    int h = threadIdx.x >> 5;   // 4 waves = 4 heads
    int src, dst;
    if (e < NEDGE) { src = ei[e]; dst = ei[NEDGE + e]; }
    else           { src = dst = e - NEDGE; }
    const float* pl = xl + (size_t)src * HID + h * 128;
    const float* pr = xr + (size_t)dst * HID + h * 128;
    const float* pa = att + h * 128;
    float s = 0.0f;
#pragma unroll
    for (int i = 0; i < 4; ++i) {
        int c = lane + 32 * i;
        float v = pl[c] + pr[c];
        v = v > 0.0f ? v : 0.2f * v;
        s += v * pa[c];
    }
#pragma unroll
    for (int m = 16; m >= 1; m >>= 1) s += __shfl_xor(s, m, 32);
    if (lane == 0) {
        elog[(size_t)e * 4 + h] = s;
        atomicMax(&emaxU[dst * 4 + h], orderKey(s));
    }
}

__global__ void edge_exp_kernel(const int* __restrict__ ei,
                                const float* __restrict__ elog,
                                const unsigned* __restrict__ emaxU,
                                float* __restrict__ eex,
                                float* __restrict__ den) {
    int idx = blockIdx.x * blockDim.x + threadIdx.x;
    if (idx >= NETOT * 4) return;
    int e = idx >> 2, h = idx & 3;
    int dst = (e < NEDGE) ? ei[NEDGE + e] : e - NEDGE;
    float ex = expf(elog[idx] - keyToFloat(emaxU[dst * 4 + h]));
    eex[idx] = ex;
    atomicAdd(&den[dst * 4 + h], ex);
}

__global__ void edge_agg_kernel(const int* __restrict__ ei,
                                const float* __restrict__ eex,
                                const float* __restrict__ den,
                                const float* __restrict__ xl,
                                float* __restrict__ out) {
    int wid = (blockIdx.x * blockDim.x + threadIdx.x) >> 5;
    int lane = threadIdx.x & 31;
    if (wid >= NETOT) return;
    int e = wid, src, dst;
    if (e < NEDGE) { src = ei[e]; dst = ei[NEDGE + e]; }
    else           { src = dst = e - NEDGE; }
    float alpha[4];
#pragma unroll
    for (int h = 0; h < 4; ++h)
        alpha[h] = eex[(size_t)e * 4 + h] / (den[dst * 4 + h] + 1e-16f);
    const float* ps = xl + (size_t)src * HID;
    float* pd = out + (size_t)dst * HID;
#pragma unroll
    for (int j = 0; j < 16; ++j) {
        int ch = lane + 32 * j;
        atomicAdd(&pd[ch], alpha[ch >> 7] * ps[ch]);
    }
}

__global__ void add_time_kernel(const float* __restrict__ h, float* __restrict__ h2t) {
    int row = blockIdx.x, tid = threadIdx.x;
    __shared__ float red[256];
    float p = 0.0f;
    for (int j = tid; j < HID; j += 256) { float v = h[(size_t)row * HID + j]; p += v * v; }
    red[tid] = p; __syncthreads();
    for (int s = 128; s > 0; s >>= 1) { if (tid < s) red[tid] += red[tid + s]; __syncthreads(); }
    if (tid == 0) h2t[(size_t)row * 513] = sqrtf(1.0f + red[0]);
    for (int j = tid; j < HID; j += 256)
        h2t[(size_t)row * 513 + 1 + j] = h[(size_t)row * HID + j];
}

__global__ void centroid_kernel(const float* __restrict__ h2t, float* __restrict__ gm) {
    int b = blockIdx.x, tid = threadIdx.x;
    __shared__ float avg[513];
    __shared__ float red[256];
    for (int col = tid; col < 513; col += 256) {
        float s = 0.0f;
        for (int i = 0; i < 128; ++i) s += h2t[((size_t)(b * 128 + i)) * 513 + col];
        avg[col] = s * (1.0f / 128.0f);
    }
    __syncthreads();
    float p = 0.0f;
    for (int col = tid; col < 513; col += 256) {
        float v = avg[col] * avg[col];
        p += (col == 0) ? -v : v;
    }
    red[tid] = p; __syncthreads();
    for (int s = 128; s > 0; s >>= 1) { if (tid < s) red[tid] += red[tid + s]; __syncthreads(); }
    float d = sqrtf(fmaxf(-red[0], 1e-8f));
    for (int col = tid; col < 513; col += 256)
        gm[(size_t)b * 513 + col] = avg[col] / d;
}

__global__ void pack_z_kernel(const float* __restrict__ h2t, __bf16* __restrict__ zbf) {
    int idx = blockIdx.x * blockDim.x + threadIdx.x;
    if (idx >= BATCH * 544) return;
    int b = idx / 544, c = idx - b * 544;
    float v = (c < 513) ? h2t[((size_t)b * 128) * 513 + c] : 0.0f;
    zbf[idx] = (__bf16)v;
}

__global__ void llin_post_kernel(const float* __restrict__ in, int inLd, int S,
                                 const float* __restrict__ sPtr,
                                 float* __restrict__ outF, int outLd,
                                 __bf16* __restrict__ outB, int bLd) {
    int b = blockIdx.x, tid = threadIdx.x;
    __shared__ float red[256];
    const float* row = in + (size_t)b * inLd;
    float t = 1.0f / (1.0f + expf(-row[0])) * expf(*sPtr) + 1.1f;
    float p = 0.0f;
    for (int j = tid; j < S; j += 256) { float v = row[1 + j]; p += v * v; }
    red[tid] = p; __syncthreads();
    for (int s = 128; s > 0; s >>= 1) { if (tid < s) red[tid] += red[tid + s]; __syncthreads(); }
    float sq  = fmaxf(red[0], 1e-8f);
    float fac = sqrtf((t * t - 1.0f) / sq);
    if (outF) {
        if (tid == 0) outF[(size_t)b * outLd] = t;
        for (int j = tid; j < S; j += 256) outF[(size_t)b * outLd + 1 + j] = row[1 + j] * fac;
    }
    if (outB) {
        if (tid == 0) outB[(size_t)b * bLd] = (__bf16)t;
        for (int j = tid; j < S; j += 256) outB[(size_t)b * bLd + 1 + j] = (__bf16)(row[1 + j] * fac);
        for (int j = S + 1 + tid; j < bLd; j += 256) outB[(size_t)b * bLd + j] = (__bf16)0.0f;
    }
}

__global__ void gelu_addtime_kernel(const float* __restrict__ z1, __bf16* __restrict__ z2b) {
    int b = blockIdx.x, tid = threadIdx.x;
    __shared__ float red[256];
    const float* row = z1 + (size_t)b * 2049 + 1;
    float p = 0.0f;
    for (int j = tid; j < 2048; j += 256) { float g = geluf(row[j]); p += g * g; }
    red[tid] = p; __syncthreads();
    for (int s = 128; s > 0; s >>= 1) { if (tid < s) red[tid] += red[tid + s]; __syncthreads(); }
    float t = sqrtf(1.0f + red[0]);
    __bf16* orow = z2b + (size_t)b * 2080;
    if (tid == 0) orow[0] = (__bf16)t;
    for (int j = tid; j < 2048; j += 256) orow[1 + j] = (__bf16)geluf(row[j]);
    for (int j = 2049 + tid; j < 2080; j += 256) orow[j] = (__bf16)0.0f;
}

extern "C" void kernel_launch(void* const* d_in, const int* in_sizes, int n_in,
                              void* d_out, int out_size, void* d_ws, size_t ws_size,
                              hipStream_t stream) {
    (void)in_sizes; (void)n_in; (void)out_size; (void)ws_size;
    const float* x    = (const float*)d_in[0];
    const int*   ei   = (const int*)d_in[1];
    const float* Wl1  = (const float*)d_in[3];
    const float* bl1  = (const float*)d_in[4];
    const float* Wr1  = (const float*)d_in[5];
    const float* br1  = (const float*)d_in[6];
    const float* att1 = (const float*)d_in[7];
    const float* bias1= (const float*)d_in[8];
    const float* Wl2  = (const float*)d_in[9];
    const float* bl2  = (const float*)d_in[10];
    const float* Wr2  = (const float*)d_in[11];
    const float* br2  = (const float*)d_in[12];
    const float* att2 = (const float*)d_in[13];
    const float* bias2= (const float*)d_in[14];
    const float* Wa   = (const float*)d_in[15];
    const float* ba   = (const float*)d_in[16];
    const float* sa   = (const float*)d_in[17];
    const float* Wb   = (const float*)d_in[18];
    const float* bb   = (const float*)d_in[19];
    const float* sb   = (const float*)d_in[20];
    const float* Wf   = (const float*)d_in[21];
    const float* bfb  = (const float*)d_in[22];
    const float* sf   = (const float*)d_in[23];
    float* out = (float*)d_out;

    char* base = (char*)d_ws; size_t off = 0;
    auto alloc = [&](size_t bytes) -> void* {
        void* p = base + off; off += (bytes + 255) & ~(size_t)255; return p;
    };
    __bf16*   hbf   = (__bf16*)  alloc((size_t)NNODE * HID * 2);
    __bf16*   wT0   = (__bf16*)  alloc((size_t)2080 * 640 * 2 + (size_t)2176 * 544 * 2); // max weight panel
    __bf16*   wT1   = (__bf16*)  alloc((size_t)512 * 512 * 2);
    float*    xl    = (float*)   alloc((size_t)NNODE * HID * 4);
    float*    xr    = (float*)   alloc((size_t)NNODE * HID * 4);
    float*    hagg  = (float*)   alloc((size_t)NNODE * HID * 4);
    float*    elog  = (float*)   alloc((size_t)NETOT * 4 * 4);
    float*    eex   = (float*)   alloc((size_t)NETOT * 4 * 4);
    unsigned* emaxU = (unsigned*)alloc((size_t)NNODE * 4 * 4);
    float*    den   = (float*)   alloc((size_t)NNODE * 4 * 4);
    float*    h2t   = (float*)   alloc((size_t)NNODE * 513 * 4);
    __bf16*   zbfA  = (__bf16*)  alloc((size_t)BATCH * 544 * 2);
    float*    za    = (float*)   alloc((size_t)BATCH * 2176 * 4);
    float*    z1    = (float*)   alloc((size_t)BATCH * 2049 * 4);
    __bf16*   z2bf  = (__bf16*)  alloc((size_t)BATCH * 2080 * 2);
    float*    zb    = (float*)   alloc((size_t)BATCH * 640 * 4);
    __bf16*   z3bf  = (__bf16*)  alloc((size_t)BATCH * 544 * 2);
    float*    zf    = (float*)   alloc((size_t)BATCH * 640 * 4);

    const int NH = NNODE * HID;
    dim3 blk(256);
    dim3 gridNH((NH + 255) / 256);
    dim3 gridGemmBig(HID / 128, NNODE / 128);   // (4, 128)

    // ---- GAT layer 1 ----
    cast_x_kernel<<<gridNH, blk, 0, stream>>>(x, hbf);
    transpose_cast_kernel<<<(512 * 512 + 255) / 256, blk, 0, stream>>>(Wl1, 512, 512, wT0, 512, 512);
    transpose_cast_kernel<<<(512 * 512 + 255) / 256, blk, 0, stream>>>(Wr1, 512, 512, wT1, 512, 512);
    gemm_bf16_kernel<<<gridGemmBig, blk, 0, stream>>>(hbf, wT0, bl1, 512, xl, 512, 512, 512, 512);
    gemm_bf16_kernel<<<gridGemmBig, blk, 0, stream>>>(hbf, wT1, br1, 512, xr, 512, 512, 512, 512);
    hipMemsetAsync(emaxU, 0, (size_t)NNODE * 4 * 4, stream);
    hipMemsetAsync(den,   0, (size_t)NNODE * 4 * 4, stream);
    bias_init_kernel<<<gridNH, blk, 0, stream>>>(bias1, hagg);
    edge_logits_kernel<<<NETOT, 128, 0, stream>>>(ei, xl, xr, att1, elog, emaxU);
    edge_exp_kernel<<<(NETOT * 4 + 255) / 256, blk, 0, stream>>>(ei, elog, emaxU, eex, den);
    edge_agg_kernel<<<NETOT / 8, blk, 0, stream>>>(ei, eex, den, xl, hagg);

    // ---- gelu + GAT layer 2 ----
    gelu_cast_kernel<<<gridNH, blk, 0, stream>>>(hagg, hbf);
    transpose_cast_kernel<<<(512 * 512 + 255) / 256, blk, 0, stream>>>(Wl2, 512, 512, wT0, 512, 512);
    transpose_cast_kernel<<<(512 * 512 + 255) / 256, blk, 0, stream>>>(Wr2, 512, 512, wT1, 512, 512);
    gemm_bf16_kernel<<<gridGemmBig, blk, 0, stream>>>(hbf, wT0, bl2, 512, xl, 512, 512, 512, 512);
    gemm_bf16_kernel<<<gridGemmBig, blk, 0, stream>>>(hbf, wT1, br2, 512, xr, 512, 512, 512, 512);
    hipMemsetAsync(emaxU, 0, (size_t)NNODE * 4 * 4, stream);
    hipMemsetAsync(den,   0, (size_t)NNODE * 4 * 4, stream);
    bias_init_kernel<<<gridNH, blk, 0, stream>>>(bias2, hagg);
    edge_logits_kernel<<<NETOT, 128, 0, stream>>>(ei, xl, xr, att2, elog, emaxU);
    edge_exp_kernel<<<(NETOT * 4 + 255) / 256, blk, 0, stream>>>(ei, elog, emaxU, eex, den);
    edge_agg_kernel<<<NETOT / 8, blk, 0, stream>>>(ei, eex, den, xl, hagg);

    // ---- add_time, centroid (gm -> second half of d_out), z extraction ----
    add_time_kernel<<<NNODE, blk, 0, stream>>>(hagg, h2t);
    centroid_kernel<<<BATCH, blk, 0, stream>>>(h2t, out + (size_t)BATCH * 513);
    pack_z_kernel<<<(BATCH * 544 + 255) / 256, blk, 0, stream>>>(h2t, zbfA);

    // ---- llin a ----
    transpose_cast_kernel<<<(2176 * 544 + 255) / 256, blk, 0, stream>>>(Wa, 513, 2049, wT0, 544, 2176);
    gemm_bf16_kernel<<<dim3(17, 1), blk, 0, stream>>>(zbfA, wT0, ba, 2049, za, 544, 544, 544, 2176);
    llin_post_kernel<<<BATCH, blk, 0, stream>>>(za, 2176, 2048, sa, z1, 2049, nullptr, 0);
    gelu_addtime_kernel<<<BATCH, blk, 0, stream>>>(z1, z2bf);

    // ---- llin b ----
    transpose_cast_kernel<<<(640 * 2080 + 255) / 256, blk, 0, stream>>>(Wb, 2049, 513, wT0, 2080, 640);
    gemm_bf16_kernel<<<dim3(5, 1), blk, 0, stream>>>(z2bf, wT0, bb, 513, zb, 2080, 2080, 2080, 640);
    llin_post_kernel<<<BATCH, blk, 0, stream>>>(zb, 640, 512, sb, nullptr, 0, z3bf, 544);

    // ---- llin f -> first half of d_out ----
    transpose_cast_kernel<<<(640 * 544 + 255) / 256, blk, 0, stream>>>(Wf, 513, 513, wT0, 544, 640);
    gemm_bf16_kernel<<<dim3(5, 1), blk, 0, stream>>>(z3bf, wT0, bfb, 513, zf, 544, 544, 544, 640);
    llin_post_kernel<<<BATCH, blk, 0, stream>>>(zf, 640, 512, sf, out, 513, nullptr, 0);
}
